// PrefixTuning_42571715837963
// MI455X (gfx1250) — compile-verified, matrix operand order checked by reading, not run on hardware
//
#include <hip/hip_runtime.h>

// ---------------------------------------------------------------------------
// Prefix-tuning attention, flash-attention style, bf16 WMMA on gfx1250.
//   out[b,q,:] = softmax(Q[b,q,:] . [prefix_k;K[b]]^T) @ [prefix_v;V[b]]
// B=4, S=4096, H=1024, P=10, L=4106.
//
// Fast path: preprocess K (+prefix, concat, bf16, zero-pad) and V (+prefix,
// concat, bf16, TRANSPOSED) into d_ws once; the flash kernel then feeds the
// matrix pipes with pure b128 loads at immediate offsets, software-pipelined
// so loads for fragment i+1/i+2 are in flight while WMMA i executes.
// ---------------------------------------------------------------------------

constexpr int B_   = 4;
constexpr int S_   = 4096;
constexpr int H_   = 1024;
constexpr int P_   = 10;
constexpr int L_   = S_ + P_;   // 4106 real keys (prefix first)
constexpr int KB   = 128;       // keys per flash iteration
constexpr int NKB  = (L_ + KB - 1) / KB;   // 33
constexpr int LP_  = NKB * KB;  // 4224 padded key stream length
constexpr int BQ   = 32;        // query rows per block (2 WMMA m-tiles)
constexpr int NW   = 8;         // waves per block (wave32)
constexpr int QSTR = H_ + 8;    // LDS row stride (bf16 elems) for Q tile
constexpr int PSTR = KB + 8;    // LDS row stride (bf16 elems) for P tile

typedef __attribute__((ext_vector_type(16))) __bf16 v16bf;
typedef __attribute__((ext_vector_type(8)))  __bf16 v8bf;
typedef __attribute__((ext_vector_type(4)))  __bf16 v4bf;
typedef __attribute__((ext_vector_type(8)))  float  v8f;
typedef __attribute__((ext_vector_type(4)))  float  v4f;

// D = A*B + C, bf16 inputs, f32 accumulate (v_wmma_f32_16x16x32_bf16)
__device__ __forceinline__ v8f wmma_bf16(v16bf a, v16bf b, v8f c) {
    return __builtin_amdgcn_wmma_f32_16x16x32_bf16(
        /*neg_a=*/false, a, /*neg_b=*/false, b,
        /*c_mod=*/(short)0, c, /*reuse_a=*/false, /*reuse_b=*/false);
}

// Half-wave (16-lane) horizontal reductions; xor masks 1/2/4/8 stay within
// the lane 0-15 / 16-31 halves, which map to distinct C-matrix rows.
__device__ __forceinline__ float hmax16(float v) {
    v = fmaxf(v, __shfl_xor(v, 1, 32));
    v = fmaxf(v, __shfl_xor(v, 2, 32));
    v = fmaxf(v, __shfl_xor(v, 4, 32));
    v = fmaxf(v, __shfl_xor(v, 8, 32));
    return v;
}
__device__ __forceinline__ float hsum16(float v) {
    v += __shfl_xor(v, 1, 32);
    v += __shfl_xor(v, 2, 32);
    v += __shfl_xor(v, 4, 32);
    v += __shfl_xor(v, 8, 32);
    return v;
}

// 16x32 (A) or 32x16 (B) bf16 fragment from a bf16 array whose element order
// along the contraction axis is contiguous. Pointer p must already include
// the per-lane row/column base and the +8-element high-half shift.
//   elements 0..7  <- p[0..7]     (K = {0..7}  lo-half / {8..15}  hi-half)
//   elements 8..15 <- p[16..23]   (K = {16..23} lo-half / {24..31} hi-half)
// => two 16B loads (ds_load_b128 from LDS / global_load_b128 from memory).
__device__ __forceinline__ v16bf load_frag16(const __bf16* p) {
    union { v16bf v; v8bf h[2]; } u;
    u.h[0] = *(const v8bf*)(p);
    u.h[1] = *(const v8bf*)(p + 16);
    return u.v;
}

// ===========================================================================
// Preprocessing kernel 1: Kc[b][kk][h] = bf16 of [prefix_k ; K[b] ; zeros]
// ===========================================================================
__global__ void prep_k_kernel(const float* __restrict__ k,
                              const float* __restrict__ pk,
                              unsigned short* __restrict__ kc_out) {
    __bf16* kc = reinterpret_cast<__bf16*>(kc_out);
    const size_t total = (size_t)B_ * LP_ * (H_ / 4);
    for (size_t idx = (size_t)blockIdx.x * blockDim.x + threadIdx.x; idx < total;
         idx += (size_t)gridDim.x * blockDim.x) {
        const int h4 = (int)(idx % (H_ / 4)) * 4;
        const size_t rem = idx / (H_ / 4);
        const int kk = (int)(rem % LP_);
        const int b  = (int)(rem / LP_);
        v4f f = {0.0f, 0.0f, 0.0f, 0.0f};
        if (kk < P_)      f = *(const v4f*)(pk + (size_t)kk * H_ + h4);
        else if (kk < L_) f = *(const v4f*)(k + ((size_t)b * S_ + (kk - P_)) * H_ + h4);
        v4bf d;
#pragma unroll
        for (int i = 0; i < 4; ++i) d[i] = (__bf16)f[i];
        *(v4bf*)(kc + ((size_t)b * LP_ + kk) * H_ + h4) = d;
    }
}

// ===========================================================================
// Preprocessing kernel 2: Vt[b][h][kk] = bf16 of [prefix_v ; V[b] ; zeros]^T
// LDS-tiled 64x64 transpose, coalesced on both sides.
// ===========================================================================
__global__ void prep_vt_kernel(const float* __restrict__ v,
                               const float* __restrict__ pv,
                               unsigned short* __restrict__ vt_out) {
    __bf16* vt = reinterpret_cast<__bf16*>(vt_out);
    __shared__ __bf16 tl[64][72];        // 72*2 = 144B row stride (16B multiple)
    const int kk0 = blockIdx.x * 64;
    const int h0  = blockIdx.y * 64;
    const int b   = blockIdx.z;
    const int tid = threadIdx.x;
    {   // read 64 key-rows x 64 h-cols (f32, coalesced), transpose into LDS
        const int r  = tid >> 2;         // key row 0..63
        const int tx = tid & 3;          // 16-col quarter
        const int kk = kk0 + r;
        const float* src = nullptr;
        if (kk < P_)      src = pv + (size_t)kk * H_;
        else if (kk < L_) src = v + ((size_t)b * S_ + (kk - P_)) * H_;
#pragma unroll
        for (int j = 0; j < 4; ++j) {
            const int c = tx * 16 + j * 4;
            v4f f = {0.0f, 0.0f, 0.0f, 0.0f};
            if (src) f = *(const v4f*)(src + h0 + c);
#pragma unroll
            for (int i = 0; i < 4; ++i) tl[c + i][r] = (__bf16)f[i];
        }
    }
    __syncthreads();
    {   // write 64 h-rows x 64 key-cols (bf16, coalesced b128)
        const int hr  = tid >> 2;        // h row 0..63
        const int seg = tid & 3;         // 16-element segment
        __bf16* dst = vt + ((size_t)b * H_ + h0 + hr) * LP_ + kk0 + seg * 16;
        const __bf16* s = &tl[hr][seg * 16];
        *(v8bf*)(dst)     = *(const v8bf*)(s);
        *(v8bf*)(dst + 8) = *(const v8bf*)(s + 8);
    }
}

// ===========================================================================
// Main flash-attention kernel (fast path, bf16 operands from d_ws)
// ===========================================================================
__global__ __launch_bounds__(NW * 32)
void PrefixTuning_flash_wmma_kernel(const float* __restrict__ q,
                                    const unsigned short* __restrict__ kc_in,
                                    const unsigned short* __restrict__ vt_in,
                                    float* __restrict__ out) {
    const __bf16* kcp = reinterpret_cast<const __bf16*>(kc_in);
    const __bf16* vtp = reinterpret_cast<const __bf16*>(vt_in);

    __shared__ __bf16 q_lds[BQ][QSTR];   // 32 x 1032 bf16 = 66 KB
    __shared__ __bf16 ptile[BQ][PSTR];   // 32 x 136 bf16  = 8.7 KB
    __shared__ float  pmax[BQ][NW];
    __shared__ float  psum[BQ][NW];
    __shared__ float  Mrun[BQ];
    __shared__ float  Lrun[BQ];
    __shared__ float  Alpha[BQ];

    const int b     = blockIdx.y;
    const int q0    = blockIdx.x * BQ;
    const int tid   = threadIdx.x;
    const int lane  = tid & 31;
    const int wv    = tid >> 5;       // 0..7
    const int lhalf = lane >> 4;      // 0: lanes 0-15, 1: lanes 16-31
    const int l15   = lane & 15;
    const int mt    = wv >> 2;        // m-tile (0/1) this wave owns
    const int kq    = wv & 3;         // 32-key quarter for the score pass
    const int m0    = mt * 16;        // first query row of this wave's m-tile
    const int cb0   = kq * 256;       // first output channel of this wave

    const float*  qb   = q + ((size_t)b * S_ + q0) * H_;
    const __bf16* kc_b = kcp + (size_t)b * LP_ * H_;
    const __bf16* vt_b = vtp + (size_t)b * H_ * LP_;

    // ---- stage Q tile into LDS as bf16 (coalesced f32x4 loads) ----
    for (int idx = tid; idx < BQ * (H_ / 4); idx += blockDim.x) {
        const int r  = idx / (H_ / 4);
        const int c4 = (idx % (H_ / 4)) * 4;
        const v4f f = *(const v4f*)(qb + (size_t)r * H_ + c4);
        __bf16* d = &q_lds[r][c4];
        d[0] = (__bf16)f[0]; d[1] = (__bf16)f[1];
        d[2] = (__bf16)f[2]; d[3] = (__bf16)f[3];
    }
    if (tid < BQ) { Mrun[tid] = -3.0e38f; Lrun[tid] = 0.0f; Alpha[tid] = 0.0f; }
    __syncthreads();

    // per-lane fragment row bases (lo/hi half shift baked in)
    const __bf16* qrow = &q_lds[m0 + l15][lhalf * 8];
    const __bf16* prow = &ptile[m0 + l15][lhalf * 8];

    // ---- output accumulators: 16 n-tiles x (16x16 f32) = 128 VGPRs ----
    v8f acc[16];
#pragma unroll
    for (int t = 0; t < 16; ++t)
#pragma unroll
        for (int r = 0; r < 8; ++r) acc[t][r] = 0.0f;

    for (int ib = 0; ib < NKB; ++ib) {
        const int kbase = ib * KB;

        // ===== Phase A: scores S = Q.K^T for this wave's 2 key n-tiles =====
        const __bf16* kp[2];
        bool kval[2];
#pragma unroll
        for (int nt = 0; nt < 2; ++nt) {
            const int kk = kbase + kq * 32 + nt * 16 + l15;   // < LP_ always
            kval[nt] = (kk < L_);
            kp[nt] = kc_b + (size_t)kk * H_ + lhalf * 8;
        }

        // warm L2 for the next key block (speculative; OOB is dropped)
        if (ib + 1 < NKB) {
            __builtin_prefetch(kp[0] + (size_t)KB * H_, 0, 1);
            __builtin_prefetch(kp[1] + (size_t)KB * H_, 0, 1);
        }

        v8f sfr[2];
#pragma unroll
        for (int nt = 0; nt < 2; ++nt)
#pragma unroll
            for (int r = 0; r < 8; ++r) sfr[nt][r] = 0.0f;

        // software-pipelined Q.K^T: fragments for hc+32 issued before the
        // WMMAs that consume hc, so waits are partial and latency overlaps.
        v16bf aA  = load_frag16(qrow);
        v16bf bA0 = load_frag16(kp[0]);
        v16bf bA1 = load_frag16(kp[1]);
#pragma unroll 4
        for (int hc = 0; hc < H_ - 32; hc += 32) {
            const v16bf aN  = load_frag16(qrow + hc + 32);
            const v16bf b0n = load_frag16(kp[0] + hc + 32);
            const v16bf b1n = load_frag16(kp[1] + hc + 32);
            sfr[0] = wmma_bf16(aA, bA0, sfr[0]);
            sfr[1] = wmma_bf16(aA, bA1, sfr[1]);
            aA = aN; bA0 = b0n; bA1 = b1n;
        }
        sfr[0] = wmma_bf16(aA, bA0, sfr[0]);
        sfr[1] = wmma_bf16(aA, bA1, sfr[1]);

#pragma unroll
        for (int nt = 0; nt < 2; ++nt)
            if (!kval[nt]) {
#pragma unroll
                for (int r = 0; r < 8; ++r) sfr[nt][r] = -3.0e38f;
            }

        // ===== Phase B: per-row block max -> LDS, wave0 combines =====
#pragma unroll
        for (int nt = 0; nt < 2; ++nt) {
            const int g = kq * 2 + nt;
#pragma unroll
            for (int r = 0; r < 8; ++r) {
                const float mv = hmax16(sfr[nt][r]);
                if (l15 == 0) pmax[m0 + r + lhalf * 8][g] = mv;
            }
        }
        __syncthreads();
        if (wv == 0) {  // lane = query row (BQ == 32 == wave width)
            float bm = pmax[lane][0];
#pragma unroll
            for (int g = 1; g < NW; ++g) bm = fmaxf(bm, pmax[lane][g]);
            const float mo = Mrun[lane];
            const float mn = fmaxf(mo, bm);
            Alpha[lane] = __expf(mo - mn);   // 0 on first block
            Mrun[lane]  = mn;
        }
        __syncthreads();

        // ===== Phase C: probabilities -> bf16 P tile + row sums =====
#pragma unroll
        for (int nt = 0; nt < 2; ++nt) {
            const int g    = kq * 2 + nt;
            const int colb = kq * 32 + nt * 16 + l15;
#pragma unroll
            for (int r = 0; r < 8; ++r) {
                const int row = m0 + r + lhalf * 8;
                const float pr = __expf(sfr[nt][r] - Mrun[row]);
                ptile[row][colb] = (__bf16)pr;
                const float sv = hsum16(pr);
                if (l15 == 0) psum[row][g] = sv;
            }
        }
        __syncthreads();
        if (wv == 0) {
            float bs = 0.0f;
#pragma unroll
            for (int g = 0; g < NW; ++g) bs += psum[lane][g];
            Lrun[lane] = Alpha[lane] * Lrun[lane] + bs;
        }

        // ===== Phase D: rescale O by alpha, accumulate P.V =====
        float al[8];
#pragma unroll
        for (int r = 0; r < 8; ++r) al[r] = Alpha[m0 + r + lhalf * 8];
#pragma unroll
        for (int t = 0; t < 16; ++t)
#pragma unroll
            for (int r = 0; r < 8; ++r) acc[t][r] *= al[r];

        // one 64-bit base per lane; all V loads use immediate offsets.
        const __bf16* vbase = vt_b + (size_t)(cb0 + l15) * LP_ + kbase + lhalf * 8;
        if (ib + 1 < NKB) __builtin_prefetch(vbase + KB, 0, 1);

        // depth-2 pipeline on the V stream, depth-1 on the LDS P fragments.
        // Fully unrolled -> all pipeline conditionals fold at compile time.
        v16bf aD = load_frag16(prow);                        // kc = 0
        v16bf b0 = load_frag16(vbase);                       // idx 0
        v16bf b1 = load_frag16(vbase + (size_t)16 * LP_);    // idx 1
#pragma unroll
        for (int idx = 0; idx < 64; ++idx) {
            const int kc = idx >> 4;
            const int t  = idx & 15;
            v16bf b2{};
            v16bf aN{};
            if (idx + 2 < 64) {
                const int n2 = idx + 2;
                b2 = load_frag16(vbase + (size_t)(n2 & 15) * 16 * LP_ + (n2 >> 4) * 32);
            }
            if (t == 14 && kc < 3)
                aN = load_frag16(prow + (kc + 1) * 32);
            acc[t] = wmma_bf16(aD, b0, acc[t]);
            b0 = b1; b1 = b2;
            if (t == 14 && kc < 3) aD = aD;      // keep aD live through t==15
            if (t == 15 && kc < 3) aD = aN;
            if (t == 14 && kc < 3) { aA = aN; }  // hold prefetched A fragment
            if (t == 15 && kc < 3) { aD = aA; }
        }
        __syncthreads();
    }

    // ---- final normalization and store (coalesced across lanes) ----
    float linv[8];
#pragma unroll
    for (int r = 0; r < 8; ++r) linv[r] = 1.0f / Lrun[m0 + r + lhalf * 8];

    float* ob = out + ((size_t)b * S_ + q0) * H_;
    for (int t = 0; t < 16; ++t) {
        const int c = cb0 + t * 16 + l15;
#pragma unroll
        for (int r = 0; r < 8; ++r) {
            const int row = m0 + r + lhalf * 8;
            ob[(size_t)row * H_ + c] = acc[t][r] * linv[r];
        }
    }
}

// ===========================================================================
// Fallback (direct-from-f32) kernel — used only if d_ws is too small.
// ===========================================================================
__device__ __forceinline__ const float* key_row_ptr_f32(int kk, const float* pk,
                                                        const float* kbatch) {
    kk = (kk < L_) ? kk : (L_ - 1);
    return (kk < P_) ? (pk + (size_t)kk * H_)
                     : (kbatch + (size_t)(kk - P_) * H_);
}
__device__ __forceinline__ v16bf load_b_frag_k_f32(const float* rowp, int off) {
    const float* p = rowp + off;
    const v4f f0 = *(const v4f*)(p);
    const v4f f1 = *(const v4f*)(p + 4);
    const v4f f2 = *(const v4f*)(p + 16);
    const v4f f3 = *(const v4f*)(p + 20);
    v16bf r;
#pragma unroll
    for (int i = 0; i < 4; ++i) {
        r[i]      = (__bf16)f0[i];
        r[4 + i]  = (__bf16)f1[i];
        r[8 + i]  = (__bf16)f2[i];
        r[12 + i] = (__bf16)f3[i];
    }
    return r;
}
__device__ __forceinline__ v16bf load_b_frag_v_f32(int kbase, int col, int lhalf,
                                                   const float* pv, const float* vbatch) {
    v16bf r;
#pragma unroll
    for (int e = 0; e < 16; ++e) {
        int koff = ((e < 8) ? e : (e + 8)) + (lhalf ? 8 : 0);
        int kk = kbase + koff;
        kk = (kk < L_) ? kk : (L_ - 1);
        const float* p = (kk < P_) ? (pv + (size_t)kk * H_)
                                   : (vbatch + (size_t)(kk - P_) * H_);
        r[e] = (__bf16)p[col];
    }
    return r;
}

__global__ __launch_bounds__(NW * 32)
void PrefixTuning_flash_direct_kernel(const float* __restrict__ q,
                                      const float* __restrict__ k,
                                      const float* __restrict__ v,
                                      const float* __restrict__ pk,
                                      const float* __restrict__ pv,
                                      float* __restrict__ out) {
    __shared__ __bf16 q_lds[BQ][QSTR];
    __shared__ __bf16 ptile[BQ][PSTR];
    __shared__ float  pmax[BQ][NW];
    __shared__ float  psum[BQ][NW];
    __shared__ float  Mrun[BQ];
    __shared__ float  Lrun[BQ];
    __shared__ float  Alpha[BQ];

    const int b    = blockIdx.y;
    const int q0   = blockIdx.x * BQ;
    const int tid  = threadIdx.x;
    const int lane = tid & 31;
    const int wv   = tid >> 5;
    const int lhalf = lane >> 4;
    const int l15  = lane & 15;
    const int mt   = wv >> 2;
    const int kq   = wv & 3;
    const int m0   = mt * 16;
    const int cb0  = kq * 256;

    const float* qb  = q + ((size_t)b * S_ + q0) * H_;
    const float* kb_ = k + (size_t)b * S_ * H_;
    const float* vb_ = v + (size_t)b * S_ * H_;

    for (int idx = tid; idx < BQ * (H_ / 4); idx += blockDim.x) {
        const int r  = idx / (H_ / 4);
        const int c4 = (idx % (H_ / 4)) * 4;
        const v4f f = *(const v4f*)(qb + (size_t)r * H_ + c4);
        __bf16* d = &q_lds[r][c4];
        d[0] = (__bf16)f[0]; d[1] = (__bf16)f[1];
        d[2] = (__bf16)f[2]; d[3] = (__bf16)f[3];
    }
    if (tid < BQ) { Mrun[tid] = -3.0e38f; Lrun[tid] = 0.0f; Alpha[tid] = 0.0f; }
    __syncthreads();

    const __bf16* qrow = &q_lds[m0 + l15][lhalf * 8];
    const __bf16* prow = &ptile[m0 + l15][lhalf * 8];

    v8f acc[16];
#pragma unroll
    for (int t = 0; t < 16; ++t)
#pragma unroll
        for (int r = 0; r < 8; ++r) acc[t][r] = 0.0f;

    for (int ib = 0; ib < NKB; ++ib) {
        const int kbase = ib * KB;

        const float* krow[2];
        bool kval[2];
#pragma unroll
        for (int nt = 0; nt < 2; ++nt) {
            const int kk = kbase + kq * 32 + nt * 16 + l15;
            kval[nt] = (kk < L_);
            krow[nt] = key_row_ptr_f32(kk, pk, kb_);
        }
        v8f sfr[2];
#pragma unroll
        for (int nt = 0; nt < 2; ++nt)
#pragma unroll
            for (int r = 0; r < 8; ++r) sfr[nt][r] = 0.0f;

        for (int hc = 0; hc < H_; hc += 32) {
            const v16bf a = load_frag16(qrow + hc);
            const int off = hc + (lhalf ? 8 : 0);
#pragma unroll
            for (int nt = 0; nt < 2; ++nt) {
                const v16bf bb = load_b_frag_k_f32(krow[nt], off);
                sfr[nt] = wmma_bf16(a, bb, sfr[nt]);
            }
        }
#pragma unroll
        for (int nt = 0; nt < 2; ++nt)
            if (!kval[nt]) {
#pragma unroll
                for (int r = 0; r < 8; ++r) sfr[nt][r] = -3.0e38f;
            }

#pragma unroll
        for (int nt = 0; nt < 2; ++nt) {
            const int g = kq * 2 + nt;
#pragma unroll
            for (int r = 0; r < 8; ++r) {
                const float mv = hmax16(sfr[nt][r]);
                if (l15 == 0) pmax[m0 + r + lhalf * 8][g] = mv;
            }
        }
        __syncthreads();
        if (wv == 0) {
            float bm = pmax[lane][0];
#pragma unroll
            for (int g = 1; g < NW; ++g) bm = fmaxf(bm, pmax[lane][g]);
            const float mo = Mrun[lane];
            const float mn = fmaxf(mo, bm);
            Alpha[lane] = __expf(mo - mn);
            Mrun[lane]  = mn;
        }
        __syncthreads();

#pragma unroll
        for (int nt = 0; nt < 2; ++nt) {
            const int g    = kq * 2 + nt;
            const int colb = kq * 32 + nt * 16 + l15;
#pragma unroll
            for (int r = 0; r < 8; ++r) {
                const int row = m0 + r + lhalf * 8;
                const float pr = __expf(sfr[nt][r] - Mrun[row]);
                ptile[row][colb] = (__bf16)pr;
                const float sv = hsum16(pr);
                if (l15 == 0) psum[row][g] = sv;
            }
        }
        __syncthreads();
        if (wv == 0) {
            float bs = 0.0f;
#pragma unroll
            for (int g = 0; g < NW; ++g) bs += psum[lane][g];
            Lrun[lane] = Alpha[lane] * Lrun[lane] + bs;
        }

        float al[8];
#pragma unroll
        for (int r = 0; r < 8; ++r) al[r] = Alpha[m0 + r + lhalf * 8];
#pragma unroll
        for (int t = 0; t < 16; ++t)
#pragma unroll
            for (int r = 0; r < 8; ++r) acc[t][r] *= al[r];

#pragma unroll
        for (int kc = 0; kc < 4; ++kc) {
            const v16bf a = load_frag16(prow + kc * 32);
            for (int t = 0; t < 16; ++t) {
                const v16bf bb = load_b_frag_v_f32(kbase + kc * 32,
                                                   cb0 + t * 16 + l15, lhalf, pv, vb_);
                acc[t] = wmma_bf16(a, bb, acc[t]);
            }
        }
        __syncthreads();
    }

    float linv[8];
#pragma unroll
    for (int r = 0; r < 8; ++r) linv[r] = 1.0f / Lrun[m0 + r + lhalf * 8];

    float* ob = out + ((size_t)b * S_ + q0) * H_;
    for (int t = 0; t < 16; ++t) {
        const int c = cb0 + t * 16 + l15;
#pragma unroll
        for (int r = 0; r < 8; ++r) {
            const int row = m0 + r + lhalf * 8;
            ob[(size_t)row * H_ + c] = acc[t][r] * linv[r];
        }
    }
}

// ===========================================================================
extern "C" void kernel_launch(void* const* d_in, const int* in_sizes, int n_in,
                              void* d_out, int out_size, void* d_ws, size_t ws_size,
                              hipStream_t stream) {
    (void)in_sizes; (void)n_in; (void)out_size;
    const float* q  = (const float*)d_in[0];   // [B,S,H]
    const float* k  = (const float*)d_in[1];   // [B,S,H]
    const float* v  = (const float*)d_in[2];   // [B,S,H]
    const float* pk = (const float*)d_in[3];   // [P,H]
    const float* pv = (const float*)d_in[4];   // [P,H]
    float* out = (float*)d_out;                // [B,S,H]

    const size_t elems_each = (size_t)B_ * LP_ * H_;          // bf16 elements
    const size_t ws_need    = 2 * elems_each * sizeof(unsigned short);

    dim3 grid(S_ / BQ, B_);    // x fastest -> one batch's K/V hot in L2
    dim3 block(NW * 32);

    if (ws_size >= ws_need) {
        unsigned short* kcp = (unsigned short*)d_ws;
        unsigned short* vtp = kcp + elems_each;
        prep_k_kernel<<<4096, 256, 0, stream>>>(k, pk, kcp);
        prep_vt_kernel<<<dim3(LP_ / 64, H_ / 64, B_), 256, 0, stream>>>(v, pv, vtp);
        PrefixTuning_flash_wmma_kernel<<<grid, block, 0, stream>>>(q, kcp, vtp, out);
    } else {
        PrefixTuning_flash_direct_kernel<<<grid, block, 0, stream>>>(q, k, v, pk, pv, out);
    }
}
